// SMMILe_27702539059222
// MI455X (gfx1250) — matrix operand order repaired and checked by reading, not compiled, not run on hardware
//
#include <hip/hip_runtime.h>
#include <hip/hip_bf16.h>
#include <stdint.h>

// ---------------------------------------------------------------------------
// SMMILe forward on gfx1250 (wave32, WMMA, TDM). Memory-bound on h (205MB):
// one fused pass converts h to bf16 and runs all h-consuming GEMMs via
// v_wmma_f32_16x16x32_bf16. Weight panels are staged into LDS by the Tensor
// Data Mover (tensor_load_to_lds, TENSORcnt) with double buffering so every
// wave in the block reuses them. Everything else is small reductions.
// ---------------------------------------------------------------------------

#define DD     512
#define DATTN  256
#define NC     4
#define KSEL   1000
#define NB_RED 512

typedef __bf16 bf16;
typedef __attribute__((ext_vector_type(16))) __bf16 v16bf;
typedef __attribute__((ext_vector_type(8)))  float  v8f;
typedef __attribute__((ext_vector_type(4)))  unsigned v4u;
typedef __attribute__((ext_vector_type(8)))  int      v8i_;
typedef __attribute__((ext_vector_type(4)))  int      v4i_;

struct Bytes32 { uint4 lo, hi; };

__device__ __forceinline__ bf16 tobf(float f) { return (bf16)f; }

__device__ __forceinline__ v16bf ldfragB(const bf16* base, int entry) {
  const uint4* p = (const uint4*)base + (size_t)entry * 2;
  Bytes32 r; r.lo = p[0]; r.hi = p[1];
  return __builtin_bit_cast(v16bf, r);
}

__device__ __forceinline__ v8f wmma_bf16(v16bf a, v16bf b, v8f c) {
  return __builtin_amdgcn_wmma_f32_16x16x32_bf16(false, a, false, b, (short)0, c,
                                                 false, false);
}

// ---- Tensor Data Mover: 1D tile (dwords x 1) from global into LDS ---------
// D# per CDNA5 ISA: group0 = {count, lds_addr, global_addr,
// type=2}; group1 = {data_size=4B, tensor_dim0=tile_dim0=dwords, dims1=1}.
__device__ __forceinline__ void tdm_load_1d(unsigned ldsAddr, const void* g,
                                            unsigned dwords) {
  unsigned long long ga = (unsigned long long)g;
  v4u g0;
  g0[0] = 1u;                                   // count=1, user descriptor
  g0[1] = ldsAddr;                              // LDS byte address
  g0[2] = (unsigned)(ga & 0xFFFFFFFFu);         // global_addr[31:0]
  g0[3] = (unsigned)((ga >> 32) & 0x01FFFFFFu)  // global_addr[56:32]
          | (2u << 30);                         // type = 2 ("image")
  v8i_ g1;
  g1[0] = (int)(2u << 16);                      // data_size=2 (4B), no mcast
  g1[1] = (int)((dwords & 0xFFFFu) << 16);      // tensor_dim0[15:0]
  g1[2] = (int)((dwords >> 16) | (1u << 16));   // tensor_dim0[31:16], dim1=1
  g1[3] = (int)((dwords & 0xFFFFu) << 16);      // tile_dim0
  g1[4] = 1;                                    // tile_dim1=1, tile_dim2=0
  g1[5] = (int)dwords;                          // tensor_dim0_stride[31:0]
  g1[6] = 0;
  g1[7] = 0;
  v4i_ z4 = {0, 0, 0, 0};
#if __clang_major__ >= 23
  v8i_ z8 = {0, 0, 0, 0, 0, 0, 0, 0};
  __builtin_amdgcn_tensor_load_to_lds(g0, g1, z4, z4, z8, 0);
#else
  __builtin_amdgcn_tensor_load_to_lds(g0, g1, z4, z4, 0);
#endif
}

// reductions across the 16 lanes of a wave32 half (matches WMMA C/D layout)
__device__ __forceinline__ float rmax16(float v) {
#pragma unroll
  for (int m = 1; m < 16; m <<= 1) v = fmaxf(v, __shfl_xor(v, m, 16));
  return v;
}
__device__ __forceinline__ float rsum16(float v) {
#pragma unroll
  for (int m = 1; m < 16; m <<= 1) v += __shfl_xor(v, m, 16);
  return v;
}

__device__ __forceinline__ float blkRedMax(float v, float* red) {
  int t = threadIdx.x; red[t] = v; __syncthreads();
  for (int s = 128; s > 0; s >>= 1) { if (t < s) red[t] = fmaxf(red[t], red[t + s]); __syncthreads(); }
  float r = red[0]; __syncthreads(); return r;
}
__device__ __forceinline__ float blkRedMin(float v, float* red) {
  int t = threadIdx.x; red[t] = v; __syncthreads();
  for (int s = 128; s > 0; s >>= 1) { if (t < s) red[t] = fminf(red[t], red[t + s]); __syncthreads(); }
  float r = red[0]; __syncthreads(); return r;
}
__device__ __forceinline__ float blkRedSum(float v, float* red) {
  int t = threadIdx.x; red[t] = v; __syncthreads();
  for (int s = 128; s > 0; s >>= 1) { if (t < s) red[t] += red[t + s]; __syncthreads(); }
  float r = red[0]; __syncthreads(); return r;
}

// ---------------------------------------------------------------------------
// Weight prep: f32 -> bf16, swizzled into B-fragment order.
// B fragment (32x16 K-major tile): lane<16 -> N=lane, K=kt*32+j (j=0..15)
//                                  lane>=16 -> N=lane-16, K=kt*32+16+j
// Entry layout: [nt][kt][lane][16 bf16] so each lane reads 32 contiguous bytes.
// ---------------------------------------------------------------------------
__global__ __launch_bounds__(256) void kPrep(
    const float* __restrict__ Wa, const float* __restrict__ Wb,
    const float* __restrict__ Wcls, const float* __restrict__ Wr,
    bf16* __restrict__ waSw, bf16* __restrict__ wbSw,
    bf16* __restrict__ wclsSw, bf16* __restrict__ wrSw) {
  int t = blockIdx.x * 256 + threadIdx.x;
  if (t < 131072) {  // Wa / Wb: 16 nt x 16 kt x 32 lane x 16 j
    int j = t & 15, lane = (t >> 4) & 31, kt = (t >> 9) & 15, nt = t >> 13;
    int n = nt * 16 + (lane & 15);
    int k = kt * 32 + ((lane < 16) ? j : 16 + j);
    waSw[t] = tobf(Wa[k * DATTN + n]);
    wbSw[t] = tobf(Wb[k * DATTN + n]);
  } else if (t < 131072 + 8192) {  // Wcls padded to 512x16
    int u = t - 131072;
    int j = u & 15, lane = (u >> 4) & 31, kt = u >> 9;
    int n = lane & 15;
    int k = kt * 32 + ((lane < 16) ? j : 16 + j);
    wclsSw[u] = (n < NC) ? tobf(Wcls[k * NC + n]) : tobf(0.f);
  } else if (t < 131072 + 8192 + 24576) {  // Wr[3] padded to 512x16 each
    int u = t - 139264;
    int r = u >> 13, v = u & 8191;
    int j = v & 15, lane = (v >> 4) & 31, kt = v >> 9;
    int n = lane & 15;
    int k = kt * 32 + ((lane < 16) ? j : 16 + j);
    wrSw[u] = (n < NC + 1) ? tobf(Wr[r * DD * (NC + 1) + k * (NC + 1) + n]) : tobf(0.f);
  }
}

// ---------------------------------------------------------------------------
// Fused WMMA pass. Block = 16 waves = 256 rows; each wave owns 16 rows.
// Dynamic LDS (128KB):
//   [0      , 32K)  nt-tile buffer 0: Wa(16K) | Wb(16K)
//   [32K    , 64K)  nt-tile buffer 1: Wa(16K) | Wb(16K)
//   [64K    , 80K)  Wcls tile (padded 512x16)
//   [80K    ,128K)  Wr tiles r=0..2 (padded 512x16 each)
// Wave 0 drives the TDM: prefetch nt+1 tiles while all waves compute nt.
// ---------------------------------------------------------------------------
__global__ __launch_bounds__(512) void kFused(
    const float* __restrict__ h,
    const bf16* __restrict__ waSw, const bf16* __restrict__ wbSw,
    const bf16* __restrict__ wclsSw,  // wrSw is contiguous after wclsSw in ws
    const float* __restrict__ ba, const float* __restrict__ bb,
    const float* __restrict__ Wc, const float* __restrict__ bc,
    const float* __restrict__ bcls, const float* __restrict__ br,
    float* __restrict__ det, float* __restrict__ cls,
    float* __restrict__ refA, float* __restrict__ refB,
    float* __restrict__ refOut, int N) {
  extern __shared__ __align__(16) char smem[];
  const int lane = threadIdx.x & 31;
  const int wid  = threadIdx.x >> 5;
  const int half = lane >> 4;
  const int n    = lane & 15;
  const int rowBase = blockIdx.x * 256 + wid * 16;
  const bool loader = (wid == 0);
  const unsigned ldsBase = (unsigned)(uintptr_t)smem;

  // Kick off TDM: skinny heads (Wcls+Wr, 64KB contiguous) + nt=0 tiles.
  if (loader) {
    tdm_load_1d(ldsBase + 65536, wclsSw, 16384);        // 64KB heads
    tdm_load_1d(ldsBase + 0,     waSw,   4096);         // Wa nt=0 (16KB)
    tdm_load_1d(ldsBase + 16384, wbSw,   4096);         // Wb nt=0 (16KB)
  }

  // Load & convert 16 rows x 512 of h into 16 A fragments (kept in VGPRs).
  const int r0 = min(rowBase + n, N - 1);
  const float4* hp = (const float4*)(h + (size_t)r0 * DD);
  v16bf afr[16];
#pragma unroll
  for (int kt = 0; kt < 16; ++kt) {
    float4 x0 = hp[kt * 8 + half * 4 + 0];
    float4 x1 = hp[kt * 8 + half * 4 + 1];
    float4 y0 = hp[kt * 8 + half * 4 + 2];
    float4 y1 = hp[kt * 8 + half * 4 + 3];
    v16bf a;
    a[0] = tobf(x0.x); a[1] = tobf(x0.y); a[2]  = tobf(x0.z); a[3]  = tobf(x0.w);
    a[4] = tobf(x1.x); a[5] = tobf(x1.y); a[6]  = tobf(x1.z); a[7]  = tobf(x1.w);
    a[8] = tobf(y0.x); a[9] = tobf(y0.y); a[10] = tobf(y0.z); a[11] = tobf(y0.w);
    a[12] = tobf(y1.x); a[13] = tobf(y1.y); a[14] = tobf(y1.z); a[15] = tobf(y1.w);
    afr[kt] = a;
  }
  if (rowBase + 256 < N)  // gfx1250 global_prefetch_b8 for next block's tile
    __builtin_prefetch(h + (size_t)(rowBase + 256 + n) * DD, 0, 1);

  if (loader) __builtin_amdgcn_s_wait_tensorcnt(0);
  __syncthreads();  // heads + nt=0 tiles resident in LDS

  float detacc[8][4];
#pragma unroll
  for (int i = 0; i < 8; i++) { detacc[i][0] = detacc[i][1] = detacc[i][2] = detacc[i][3] = 0.f; }

  // Gated attention: a=tanh(h@Wa+ba), b=sigmoid(h@Wb+bb), det += (a*b)@Wc
  for (int nt = 0; nt < 16; ++nt) {
    const int cur = nt & 1;
    if (loader && nt + 1 < 16) {  // prefetch next nt tiles via TDM
      tdm_load_1d(ldsBase + (1 - cur) * 32768,         waSw + (nt + 1) * 8192, 4096);
      tdm_load_1d(ldsBase + (1 - cur) * 32768 + 16384, wbSw + (nt + 1) * 8192, 4096);
    }
    const bf16* sWa = (const bf16*)(smem + cur * 32768);
    const bf16* sWb = (const bf16*)(smem + cur * 32768 + 16384);
    v8f aacc = {0, 0, 0, 0, 0, 0, 0, 0};
    v8f bacc = {0, 0, 0, 0, 0, 0, 0, 0};
#pragma unroll
    for (int kt = 0; kt < 16; ++kt)
      aacc = wmma_bf16(afr[kt], ldfragB(sWa, kt * 32 + lane), aacc);
#pragma unroll
    for (int kt = 0; kt < 16; ++kt)
      bacc = wmma_bf16(afr[kt], ldfragB(sWb, kt * 32 + lane), bacc);
    float ban = ba[nt * 16 + n];
    float bbn = bb[nt * 16 + n];
    float4 wc4 = ((const float4*)Wc)[nt * 16 + n];
#pragma unroll
    for (int i = 0; i < 8; i++) {
      float av = tanhf(aacc[i] + ban);
      float sv = 1.f / (1.f + __expf(-(bacc[i] + bbn)));
      float g = av * sv;
      detacc[i][0] += g * wc4.x; detacc[i][1] += g * wc4.y;
      detacc[i][2] += g * wc4.z; detacc[i][3] += g * wc4.w;
    }
    __syncthreads();                                   // done reading buf cur
    if (loader) __builtin_amdgcn_s_wait_tensorcnt(0);  // next tiles landed
    __syncthreads();
  }
#pragma unroll
  for (int i = 0; i < 8; i++) {
    int row = rowBase + half * 8 + i;
#pragma unroll
    for (int c = 0; c < 4; c++) {
      float s = rsum16(detacc[i][c]);
      if (n == 0 && row < N) det[(size_t)row * 4 + c] = s + bc[c];
    }
  }

  // Classifier head: softmax(h@Wcls+bcls) over 4 classes (padded to 16)
  {
    const bf16* sCls = (const bf16*)(smem + 65536);
    v8f cacc = {0, 0, 0, 0, 0, 0, 0, 0};
#pragma unroll
    for (int kt = 0; kt < 16; ++kt)
      cacc = wmma_bf16(afr[kt], ldfragB(sCls, kt * 32 + lane), cacc);
#pragma unroll
    for (int i = 0; i < 8; i++) {
      bool ok = n < NC;
      float x = ok ? (cacc[i] + bcls[n]) : -3.0e38f;
      float mx = rmax16(x);
      float e = ok ? __expf(x - mx) : 0.f;
      float s = rsum16(e);
      int row = rowBase + half * 8 + i;
      if (ok && row < N) cls[(size_t)row * 4 + n] = e / s;
    }
  }

  // Refinement heads: softmax(h@Wr[r]+br[r]) over 5 classes (padded to 16)
#pragma unroll
  for (int r = 0; r < 3; r++) {
    const bf16* sWrr = (const bf16*)(smem + 65536 + 16384 + r * 16384);
    v8f racc = {0, 0, 0, 0, 0, 0, 0, 0};
#pragma unroll
    for (int kt = 0; kt < 16; ++kt)
      racc = wmma_bf16(afr[kt], ldfragB(sWrr, kt * 32 + lane), racc);
    float* rout = (r == 0) ? refA : (r == 1) ? refB : refOut;
#pragma unroll
    for (int i = 0; i < 8; i++) {
      bool ok = n < NC + 1;
      float x = ok ? (racc[i] + br[r * 5 + n]) : -3.0e38f;
      float mx = rmax16(x);
      float e = ok ? __expf(x - mx) : 0.f;
      float s = rsum16(e);
      int row = rowBase + half * 8 + i;
      if (ok && row < N) rout[(size_t)row * 5 + n] = e / s;
    }
  }
}

// --------------------- instance softmax over N (per class) -----------------
__global__ __launch_bounds__(256) void kDetMax(const float* __restrict__ det,
                                               float* __restrict__ pmax, int N) {
  __shared__ float red[256];
  float v[4] = {-3.0e38f, -3.0e38f, -3.0e38f, -3.0e38f};
  for (int i = blockIdx.x * 256 + threadIdx.x; i < N; i += gridDim.x * 256) {
    float4 d = ((const float4*)det)[i];
    v[0] = fmaxf(v[0], d.x); v[1] = fmaxf(v[1], d.y);
    v[2] = fmaxf(v[2], d.z); v[3] = fmaxf(v[3], d.w);
  }
  for (int c = 0; c < 4; c++) {
    float r = blkRedMax(v[c], red);
    if (threadIdx.x == 0) pmax[blockIdx.x * 4 + c] = r;
  }
}
__global__ __launch_bounds__(256) void kRed4Max(const float* __restrict__ p,
                                                float* __restrict__ g, int nb) {
  __shared__ float red[256];
  for (int c = 0; c < 4; c++) {
    float v = -3.0e38f;
    for (int j = threadIdx.x; j < nb; j += 256) v = fmaxf(v, p[j * 4 + c]);
    float r = blkRedMax(v, red);
    if (threadIdx.x == 0) g[c] = r;
  }
}
__global__ __launch_bounds__(256) void kDetSum(const float* __restrict__ det,
                                               const float* __restrict__ gmax,
                                               float* __restrict__ psum, int N) {
  __shared__ float red[256];
  float m0 = gmax[0], m1 = gmax[1], m2 = gmax[2], m3 = gmax[3];
  float v[4] = {0, 0, 0, 0};
  for (int i = blockIdx.x * 256 + threadIdx.x; i < N; i += gridDim.x * 256) {
    float4 d = ((const float4*)det)[i];
    v[0] += __expf(d.x - m0); v[1] += __expf(d.y - m1);
    v[2] += __expf(d.z - m2); v[3] += __expf(d.w - m3);
  }
  for (int c = 0; c < 4; c++) {
    float r = blkRedSum(v[c], red);
    if (threadIdx.x == 0) psum[blockIdx.x * 4 + c] = r;
  }
}
__global__ __launch_bounds__(256) void kRed4Sum(const float* __restrict__ p,
                                                float* __restrict__ g, int nb) {
  __shared__ float red[256];
  for (int c = 0; c < 4; c++) {
    float v = 0.f;
    for (int j = threadIdx.x; j < nb; j += 256) v += p[j * 4 + c];
    float r = blkRedSum(v, red);
    if (threadIdx.x == 0) g[c] = r;
  }
}

// ---- final_score = cls * softmax0(det); per-class min/max/colsum; row mean --
__global__ __launch_bounds__(256) void kFinal(
    const float* __restrict__ det, const float* __restrict__ cls,
    const float* __restrict__ gmax, const float* __restrict__ gsum,
    float* __restrict__ fin, float* __restrict__ meanv,
    float* __restrict__ pfmin, float* __restrict__ pfmax,
    float* __restrict__ pys, int N) {
  __shared__ float red[256];
  float m0 = gmax[0], m1 = gmax[1], m2 = gmax[2], m3 = gmax[3];
  float i0 = 1.f / gsum[0], i1 = 1.f / gsum[1], i2 = 1.f / gsum[2], i3 = 1.f / gsum[3];
  float mn[4] = {3.0e38f, 3.0e38f, 3.0e38f, 3.0e38f};
  float mx[4] = {-3.0e38f, -3.0e38f, -3.0e38f, -3.0e38f};
  float sm[4] = {0, 0, 0, 0};
  for (int i = blockIdx.x * 256 + threadIdx.x; i < N; i += gridDim.x * 256) {
    float4 d = ((const float4*)det)[i];
    float4 c = ((const float4*)cls)[i];
    float f0 = c.x * __expf(d.x - m0) * i0;
    float f1 = c.y * __expf(d.y - m1) * i1;
    float f2 = c.z * __expf(d.z - m2) * i2;
    float f3 = c.w * __expf(d.w - m3) * i3;
    ((float4*)fin)[i] = make_float4(f0, f1, f2, f3);
    meanv[i] = 0.25f * (f0 + f1 + f2 + f3);
    mn[0] = fminf(mn[0], f0); mx[0] = fmaxf(mx[0], f0); sm[0] += f0;
    mn[1] = fminf(mn[1], f1); mx[1] = fmaxf(mx[1], f1); sm[1] += f1;
    mn[2] = fminf(mn[2], f2); mx[2] = fmaxf(mx[2], f2); sm[2] += f2;
    mn[3] = fminf(mn[3], f3); mx[3] = fmaxf(mx[3], f3); sm[3] += f3;
  }
  for (int c = 0; c < 4; c++) {
    float a = blkRedMin(mn[c], red);
    float b = blkRedMax(mx[c], red);
    float s = blkRedSum(sm[c], red);
    if (threadIdx.x == 0) {
      pfmin[blockIdx.x * 4 + c] = a;
      pfmax[blockIdx.x * 4 + c] = b;
      pys[blockIdx.x * 4 + c] = s;
    }
  }
}

__global__ __launch_bounds__(256) void kStats(
    const float* __restrict__ pfmin, const float* __restrict__ pfmax,
    const float* __restrict__ pys, const int* __restrict__ label,
    float* __restrict__ fminv, float* __restrict__ fmaxv,
    int* __restrict__ inlabel, float* __restrict__ yprob,
    float* __restrict__ yhat, int nb) {
  __shared__ float red[256];
  __shared__ float ysv[4];
  int tid = threadIdx.x;
  for (int c = 0; c < 4; c++) {
    float v = 3.0e38f;
    for (int j = tid; j < nb; j += 256) v = fminf(v, pfmin[j * 4 + c]);
    v = blkRedMin(v, red);
    if (tid == 0) fminv[c] = v;
    float w = -3.0e38f;
    for (int j = tid; j < nb; j += 256) w = fmaxf(w, pfmax[j * 4 + c]);
    w = blkRedMax(w, red);
    if (tid == 0) fmaxv[c] = w;
    float s = 0.f;
    for (int j = tid; j < nb; j += 256) s += pys[j * 4 + c];
    s = blkRedSum(s, red);
    if (tid == 0) {
      float yp = fminf(fmaxf(s, 1e-10f), 1.f - 1e-10f);
      yprob[c] = yp; ysv[c] = yp;
    }
  }
  __syncthreads();
  if (tid == 0) {
    int best = 0; float bv = ysv[0];
    for (int c = 1; c < 4; c++) if (ysv[c] > bv) { bv = ysv[c]; best = c; }
    yhat[0] = (float)best;
    int l0 = label[0], l1 = label[1];
    for (int c = 0; c < 4; c++) inlabel[c] = (l0 == c || l1 == c) ? 1 : 0;
  }
}

// ------------------------- radix top-k selection ---------------------------
__device__ __forceinline__ unsigned enc(float f) {
  unsigned u = __float_as_uint(f);
  return (u & 0x80000000u) ? ~u : (u | 0x80000000u);
}
__device__ __forceinline__ float selKey(int s, int i, const float* fin,
                                        const float* meanv, const float* ra,
                                        const float* rb) {
  if (s < 4) return fin[(size_t)i * 4 + s];   // monotone w/ min-max normalized
  if (s == 4) return -meanv[i];               // bottom-k of mean score
  if (s < 10) return ra[(size_t)i * 5 + (s - 5)];
  return rb[(size_t)i * 5 + (s - 10)];
}

__global__ __launch_bounds__(1024) void kSelect(
    const float* __restrict__ fin, const float* __restrict__ meanv,
    const float* __restrict__ refA, const float* __restrict__ refB,
    int* __restrict__ selidx, int N) {
  __shared__ unsigned hist[256];
  __shared__ unsigned sPrefix, sMask, sKK;
  __shared__ int cGT, cEQ;
  int sel = blockIdx.x, tid = threadIdx.x;
  if (tid == 0) { sPrefix = 0u; sMask = 0u; sKK = KSEL; }
  __syncthreads();
  for (int pass = 0; pass < 4; ++pass) {
    int shift = 24 - 8 * pass;
    if (tid < 256) hist[tid] = 0u;
    __syncthreads();
    unsigned pref = sPrefix, msk = sMask;
    for (int i = tid; i < N; i += 1024) {
      unsigned u = enc(selKey(sel, i, fin, meanv, refA, refB));
      if ((u & msk) == pref) atomicAdd(&hist[(u >> shift) & 255], 1u);
    }
    __syncthreads();
    if (tid == 0) {
      unsigned kk = sKK;
      int d = 255;
      for (; d > 0; --d) { unsigned c = hist[d]; if (kk <= c) break; kk -= c; }
      unsigned cd = hist[d]; if (kk > cd) kk = cd;  // safety clamp
      sPrefix = pref | ((unsigned)d << shift);
      sMask = msk | (0xFFu << shift);
      sKK = kk;
    }
    __syncthreads();
  }
  unsigned T = sPrefix;
  int kk = (int)sKK;
  if (tid == 0) { cGT = 0; cEQ = 0; }
  __syncthreads();
  int baseGT = KSEL - kk;
  for (int i = tid; i < N; i += 1024) {
    unsigned u = enc(selKey(sel, i, fin, meanv, refA, refB));
    if (u > T) {
      int p = atomicAdd(&cGT, 1);
      if (p < baseGT) selidx[sel * KSEL + p] = i;
    } else if (u == T) {
      int p = atomicAdd(&cEQ, 1);
      if (p < kk) selidx[sel * KSEL + baseGT + p] = i;
    }
  }
}

// --------------- candidate CE: logits = h[idx]@Wr[r] + br[r] ---------------
__global__ __launch_bounds__(256) void kCand(
    const int* __restrict__ selidx, const float* __restrict__ h,
    const float* __restrict__ Wr, const float* __restrict__ br,
    const float* __restrict__ fin, const float* __restrict__ meanv,
    const float* __restrict__ refA, const float* __restrict__ refB,
    const float* __restrict__ fminv, const float* __restrict__ fmaxv,
    const int* __restrict__ inlabel,
    float* __restrict__ ceBuf, float* __restrict__ cntBuf) {
  __shared__ float sWr[DD * 5];
  __shared__ float sbr[5];
  int sel = blockIdx.x / 25;
  int chunk = blockIdx.x % 25;
  int r = (sel < 5) ? 0 : ((sel < 10) ? 1 : 2);
  for (int t = threadIdx.x; t < DD * 5; t += 256) sWr[t] = Wr[r * DD * 5 + t];
  if (threadIdx.x < 5) sbr[threadIdx.x] = br[r * 5 + threadIdx.x];
  __syncthreads();
  int wid = threadIdx.x >> 5, lane = threadIdx.x & 31;
  for (int it = 0; it < 5; ++it) {
    int cand = chunk * 40 + it * 8 + wid;
    int idx = selidx[sel * KSEL + cand];
    const float* hr = h + (size_t)idx * DD;
    float a0 = 0, a1 = 0, a2 = 0, a3 = 0, a4 = 0;
#pragma unroll
    for (int kk = 0; kk < 16; ++kk) {
      int k = lane * 16 + kk;
      float x = hr[k];
      const float* w = &sWr[k * 5];
      a0 += x * w[0]; a1 += x * w[1]; a2 += x * w[2]; a3 += x * w[3]; a4 += x * w[4];
    }
#pragma unroll
    for (int m = 1; m < 32; m <<= 1) {
      a0 += __shfl_xor(a0, m, 32); a1 += __shfl_xor(a1, m, 32);
      a2 += __shfl_xor(a2, m, 32); a3 += __shfl_xor(a3, m, 32);
      a4 += __shfl_xor(a4, m, 32);
    }
    if (lane == 0) {
      float l[5] = {a0 + sbr[0], a1 + sbr[1], a2 + sbr[2], a3 + sbr[3], a4 + sbr[4]};
      int tgt = (sel < 4) ? sel : (sel == 4) ? 4 : (sel < 10) ? (sel - 5) : (sel - 10);
      float mxv = l[0];
      for (int e = 1; e < 5; e++) mxv = fmaxf(mxv, l[e]);
      float s = 0.f;
      for (int e = 0; e < 5; e++) s += __expf(l[e] - mxv);
      float ce = -(l[tgt] - mxv - __logf(s));
      bool valid;
      if (sel < 4) {
        int c = sel;
        float f = fin[(size_t)idx * 4 + c];
        float nv = (f - fminv[c]) / (fmaxv[c] - fminv[c]);
        valid = (nv > 0.5f) && (inlabel[c] != 0);
      } else if (sel == 4) {
        valid = meanv[idx] < 0.5f;
      } else if (sel < 10) {
        int c = sel - 5;
        float v = refA[(size_t)idx * 5 + c];
        valid = (v > 0.5f) && (c == 4 || inlabel[c] != 0);
      } else {
        int c = sel - 10;
        float v = refB[(size_t)idx * 5 + c];
        valid = (v > 0.5f) && (c == 4 || inlabel[c] != 0);
      }
      ceBuf[sel * KSEL + cand] = valid ? ce : 0.f;
      cntBuf[sel * KSEL + cand] = valid ? 1.f : 0.f;
    }
  }
}

__global__ __launch_bounds__(256) void kLoss(const float* __restrict__ ceBuf,
                                             const float* __restrict__ cntBuf,
                                             float* __restrict__ lossOut) {
  __shared__ float sn[256], sd[256];
  int tid = threadIdx.x;
  float loss = 0.f;
  for (int r = 0; r < 3; r++) {
    float n = 0.f, d = 0.f;
    for (int j = tid; j < 5 * KSEL; j += 256) {
      n += ceBuf[r * 5 * KSEL + j];
      d += cntBuf[r * 5 * KSEL + j];
    }
    sn[tid] = n; sd[tid] = d;
    __syncthreads();
    for (int s = 128; s > 0; s >>= 1) {
      if (tid < s) { sn[tid] += sn[tid + s]; sd[tid] += sd[tid + s]; }
      __syncthreads();
    }
    if (tid == 0) loss += sn[0] / fmaxf(sd[0], 1.f);
    __syncthreads();
  }
  if (tid == 0) lossOut[0] = loss;
}

// ---------------------------------------------------------------------------
extern "C" void kernel_launch(void* const* d_in, const int* in_sizes, int n_in,
                              void* d_out, int out_size, void* d_ws, size_t ws_size,
                              hipStream_t stream) {
  const float* h    = (const float*)d_in[0];
  const int*   lab  = (const int*)d_in[1];
  const float* Wa   = (const float*)d_in[2];
  const float* ba   = (const float*)d_in[3];
  const float* Wb   = (const float*)d_in[4];
  const float* bb   = (const float*)d_in[5];
  const float* Wc   = (const float*)d_in[6];
  const float* bc   = (const float*)d_in[7];
  const float* Wcls = (const float*)d_in[8];
  const float* bcls = (const float*)d_in[9];
  const float* Wr   = (const float*)d_in[10];
  const float* br   = (const float*)d_in[11];
  const int N = in_sizes[0] / DD;

  char* w = (char*)d_ws;
  size_t off = 0;
  auto take = [&](size_t bytes) -> char* {
    char* p = w + off;
    off = (off + bytes + 255) & ~(size_t)255;
    return p;
  };
  bf16*  waSw   = (bf16*)take((size_t)DD * DATTN * 2);
  bf16*  wbSw   = (bf16*)take((size_t)DD * DATTN * 2);
  bf16*  wclsSw = (bf16*)take((size_t)DD * 16 * 2);   // contiguous with wrSw
  bf16*  wrSw   = (bf16*)take((size_t)3 * DD * 16 * 2);
  float* det    = (float*)take((size_t)N * 4 * 4);
  float* cls    = (float*)take((size_t)N * 4 * 4);
  float* refA   = (float*)take((size_t)N * 5 * 4);
  float* refB   = (float*)take((size_t)N * 5 * 4);
  float* meanv  = (float*)take((size_t)N * 4);
  float* pmax   = (float*)take((size_t)NB_RED * 4 * 4);
  float* psum   = (float*)take((size_t)NB_RED * 4 * 4);
  float* pfmin  = (float*)take((size_t)NB_RED * 4 * 4);
  float* pfmax  = (float*)take((size_t)NB_RED * 4 * 4);
  float* pys    = (float*)take((size_t)NB_RED * 4 * 4);
  float* gmax   = (float*)take(4 * 4);
  float* gsum   = (float*)take(4 * 4);
  float* fminv  = (float*)take(4 * 4);
  float* fmaxv  = (float*)take(4 * 4);
  int*   inlab  = (int*)take(4 * 4);
  int*   selidx = (int*)take((size_t)15 * KSEL * 4);
  float* ceBuf  = (float*)take((size_t)15 * KSEL * 4);
  float* cntBuf = (float*)take((size_t)15 * KSEL * 4);

  float* out    = (float*)d_out;
  float* fin    = out;                       // final_score [N,4]
  float* yprob  = out + (size_t)N * 4;       // Y_prob [4]
  float* yhat   = yprob + 4;                 // Y_hat [1]
  float* refOut = yhat + 1;                  // ref_score [N,5]
  float* lossO  = refOut + (size_t)N * 5;    // instance_loss [1]

  kPrep<<<640, 256, 0, stream>>>(Wa, Wb, Wcls, Wr, waSw, wbSw, wclsSw, wrSw);
  kFused<<<(N + 255) / 256, 512, 131072, stream>>>(h, waSw, wbSw, wclsSw,
                                                   ba, bb, Wc, bc, bcls, br,
                                                   det, cls, refA, refB, refOut, N);
  kDetMax<<<NB_RED, 256, 0, stream>>>(det, pmax, N);
  kRed4Max<<<1, 256, 0, stream>>>(pmax, gmax, NB_RED);
  kDetSum<<<NB_RED, 256, 0, stream>>>(det, gmax, psum, N);
  kRed4Sum<<<1, 256, 0, stream>>>(psum, gsum, NB_RED);
  kFinal<<<NB_RED, 256, 0, stream>>>(det, cls, gmax, gsum, fin, meanv,
                                     pfmin, pfmax, pys, N);
  kStats<<<1, 256, 0, stream>>>(pfmin, pfmax, pys, lab, fminv, fmaxv, inlab,
                                yprob, yhat, NB_RED);
  kSelect<<<15, 1024, 0, stream>>>(fin, meanv, refA, refB, selidx, N);
  kCand<<<375, 256, 0, stream>>>(selidx, h, Wr, br, fin, meanv, refA, refB,
                                 fminv, fmaxv, inlab, ceBuf, cntBuf);
  kLoss<<<1, 256, 0, stream>>>(ceBuf, cntBuf, lossO);
}
